// CategoryHead_74912819577107
// MI455X (gfx1250) — matrix operand order repaired and checked by reading, not compiled, not run on hardware
//
#include <hip/hip_runtime.h>

typedef __attribute__((ext_vector_type(16))) __bf16 v16bf;
typedef __attribute__((ext_vector_type(8)))  __bf16 v8bf;
typedef __attribute__((ext_vector_type(4)))  __bf16 v4bf;
typedef __attribute__((ext_vector_type(8)))  float  v8f;
typedef __attribute__((ext_vector_type(4)))  float  v4f;

namespace {
constexpr int B_TOT = 65536;
constexpr int T = 2, D = 128, DFF = 512, L = 4, NC = 7;
constexpr int SPW = 16;            // samples per wave
constexpr int WPB = 2;             // waves per block (wave32)
// LDS row strides, padded for 16B alignment / bank spread
constexpr int SF  = 132;           // tok rows, f32
constexpr int TBS = 136;           // tok rows, bf16 shadow
constexpr int QS  = 392;           // qkv rows, bf16
constexpr int HS  = 520;           // ffn-h rows, bf16
constexpr int OS  = 136;           // attn-o rows, bf16
constexpr int XS  = 272;           // x staging rows, bf16
constexpr int TOK_BYTES     = 32 * SF * 4;                 // 16896
constexpr int TOKB_BYTES    = 32 * TBS * 2;                // 8704
constexpr int SCRATCH_BYTES = 32 * QS * 2 + 32 * OS * 2;   // 33792 (covers h=33280, x=8704, pooled=8192)
constexpr int WAVE_LDS      = TOK_BYTES + TOKB_BYTES + SCRATCH_BYTES;  // 59392
// bf16 weight offsets (elements) in d_ws
constexpr int OFF_TP  = 0;
constexpr int OFF_QKV = OFF_TP  + 256 * 256;
constexpr int OFF_OUT = OFF_QKV + L * 3 * D * D;
constexpr int OFF_FF1 = OFF_OUT + L * D * D;
constexpr int OFF_FF2 = OFF_FF1 + L * DFF * D;
}

// ---------- WMMA helpers (CDNA5 wave32, 16x16x32 bf16 -> f32) ----------

static __device__ __forceinline__ v8f wmma_bf16(v16bf a, v16bf b, v8f c) {
  return __builtin_amdgcn_wmma_f32_16x16x32_bf16(
      /*neg_a=*/false, a, /*neg_b=*/false, b,
      /*c_mod=*/(short)0, c, /*reuse_a=*/false, /*reuse_b=*/false);
}

// A fragment (16x32) from a bf16 row-major LDS tile.
// lane<16: row=lane,    K = kb..kb+15
// lane>=16: row=lane-16, K = kb+16..kb+31
static __device__ __forceinline__ v16bf loadA_bf16(const __bf16* tile, int ldr,
                                                   int kb, int lane) {
  const int m  = lane & 15;
  const int k0 = kb + ((lane & 16) ? 16 : 0);
  const __bf16* p = tile + m * ldr + k0;
  v8bf lo = *(const v8bf*)(p);
  v8bf hi = *(const v8bf*)(p + 8);
  v16bf r;
#pragma unroll
  for (int j = 0; j < 8; ++j) { r[j] = lo[j]; r[8 + j] = hi[j]; }
  return r;
}

// B fragment (32x16) from bf16 weights stored [N_out, K] row-major in global.
// B[k][n] = W[n][k] -> lane holds column n = n0+(lane&15), 16 contiguous K.
static __device__ __forceinline__ v16bf loadB(const __bf16* W, int ldk,
                                              int n0, int kb, int lane) {
  const int n  = n0 + (lane & 15);
  const int kk = kb + ((lane & 16) ? 16 : 0);
  return *(const v16bf*)(W + n * ldk + kk);
}

static __device__ __forceinline__ float gelu_exact(float v) {
  return 0.5f * v * (1.0f + erff(v * 0.70710678118654752f));
}

// Per-lane LayerNorm over one 128-wide f32 row; also refresh bf16 shadow row.
static __device__ __forceinline__ void ln_row(float* row, __bf16* rowb,
                                              const float* w, const float* b) {
  float s = 0.f;
#pragma unroll 8
  for (int d = 0; d < D; ++d) s += row[d];
  const float mean = s * (1.0f / D);
  float v = 0.f;
#pragma unroll 8
  for (int d = 0; d < D; ++d) { float t = row[d] - mean; v += t * t; }
  const float inv = rsqrtf(v * (1.0f / D) + 1e-5f);
#pragma unroll 8
  for (int d = 0; d < D; ++d) {
    float o = (row[d] - mean) * inv * w[d] + b[d];
    row[d]  = o;
    rowb[d] = (__bf16)o;
  }
}

// ---------- weight conversion f32 -> bf16 into d_ws ----------

__global__ void cvt_f32_bf16(const float* __restrict__ s, __bf16* __restrict__ d, int n) {
  int i = blockIdx.x * blockDim.x + threadIdx.x;
  if (i < n) d[i] = (__bf16)s[i];
}

// ---------- fully fused transformer encoder + classifier ----------

__global__ __launch_bounds__(WPB * 32) void vit_fused_kernel(
    const float* __restrict__ x,
    const __bf16* __restrict__ wTp,  const __bf16* __restrict__ wQkv,
    const __bf16* __restrict__ wOut, const __bf16* __restrict__ wFf1,
    const __bf16* __restrict__ wFf2,
    const float* __restrict__ tpB,  const float* __restrict__ pos,
    const float* __restrict__ qkvB, const float* __restrict__ outB,
    const float* __restrict__ ln1w, const float* __restrict__ ln1b,
    const float* __restrict__ ff1B, const float* __restrict__ ff2B,
    const float* __restrict__ ln2w, const float* __restrict__ ln2b,
    const float* __restrict__ lnfw, const float* __restrict__ lnfb,
    const float* __restrict__ clnw, const float* __restrict__ clnb,
    const float* __restrict__ clsW, const float* __restrict__ clsB,
    float* __restrict__ out) {
  extern __shared__ char smem[];
  const int lane = threadIdx.x & 31;
  const int wv   = threadIdx.x >> 5;
  char* wbase   = smem + wv * WAVE_LDS;
  float*  tokf  = (float*)wbase;                          // [32][SF]  f32
  __bf16* tokb  = (__bf16*)(wbase + TOK_BYTES);           // [32][TBS] bf16 shadow
  char*   scr   = wbase + TOK_BYTES + TOKB_BYTES;         // time-multiplexed
  __bf16* qkvb  = (__bf16*)scr;                           // [32][QS]
  __bf16* obuf  = (__bf16*)(scr + 32 * QS * 2);           // [32][OS]
  __bf16* hbuf  = (__bf16*)scr;                           // [32][HS] (after attn)
  __bf16* xstg  = (__bf16*)scr;                           // [16][XS] (token_proj)
  float*  pooled = (float*)scr;                           // [16][128] (head)

  const int s0 = (blockIdx.x * WPB + wv) * SPW;           // first sample of wave
  const int rb = (lane & 16) ? 8 : 0;                     // C-tile row base
  const int nl = lane & 15;                               // C-tile column lane

  // ---- stage x tile [16][256] into LDS as bf16 (one-time conversion) ----
  {
    const float* xs = x + (size_t)s0 * 256;
#pragma unroll
    for (int idx = lane; idx < 16 * 64; idx += 32) {
      int r = idx >> 6, c = (idx & 63) << 2;
      v4f v = *(const v4f*)(xs + r * 256 + c);
      v4bf o;
      o[0] = (__bf16)v[0]; o[1] = (__bf16)v[1];
      o[2] = (__bf16)v[2]; o[3] = (__bf16)v[3];
      *(v4bf*)(xstg + r * XS + c) = o;
    }
  }
  __syncthreads();

  // ---- token projection: [16,256] @ W^T(256,256) + b + pos -> tok[32][128] ----
  {
    v16bf ax[8];
#pragma unroll
    for (int k8 = 0; k8 < 8; ++k8) ax[k8] = loadA_bf16(xstg, XS, k8 * 32, lane);
    for (int nt = 0; nt < 16; ++nt) {
      v8f acc = {};
#pragma unroll
      for (int k8 = 0; k8 < 8; ++k8)
        acc = wmma_bf16(ax[k8], loadB(wTp, 256, nt * 16, k8 * 32, lane), acc);
      const int n = nt * 16 + nl;                         // 0..255
      const float bias = tpB[n] + pos[n];                 // pos_emb flat [T*D]
#pragma unroll
      for (int r = 0; r < 8; ++r) {
        int m = rb + r;                                   // sample within wave
        float val = acc[r] + bias;
        int row = m * 2 + (n >> 7), col = n & 127;
        tokf[row * SF + col]  = val;
        tokb[row * TBS + col] = (__bf16)val;
      }
    }
  }
  __syncthreads();

  // ---- encoder layers ----
  for (int li = 0; li < L; ++li) {
    const __bf16* qW  = wQkv + li * 3 * D * D;
    const __bf16* oW  = wOut + li * D * D;
    const __bf16* f1W = wFf1 + li * DFF * D;
    const __bf16* f2W = wFf2 + li * D * DFF;
    __builtin_prefetch(qW, 0, 1);
    __builtin_prefetch(f1W, 0, 1);

    // QKV: tok[32,128] @ qkv_w^T(384,128) + b -> qkvb[32][384] (bf16)
    for (int rt = 0; rt < 2; ++rt) {
      v16bf a4[4];
#pragma unroll
      for (int k4 = 0; k4 < 4; ++k4)
        a4[k4] = loadA_bf16(tokb + rt * 16 * TBS, TBS, k4 * 32, lane);
      for (int nt = 0; nt < 24; ++nt) {
        v8f acc = {};
#pragma unroll
        for (int k4 = 0; k4 < 4; ++k4)
          acc = wmma_bf16(a4[k4], loadB(qW, D, nt * 16, k4 * 32, lane), acc);
        const int n = nt * 16 + nl;
        const float bias = qkvB[li * 384 + n];
#pragma unroll
        for (int r = 0; r < 8; ++r)
          qkvb[(rt * 16 + rb + r) * QS + n] = (__bf16)(acc[r] + bias);
      }
    }
    __syncthreads();

    // attention: lane <-> row (b,t1); T=2, H=8, DH=16; f32 math
    {
      const int r1 = lane, sidx = lane >> 1;
      const int rk0 = sidx * 2, rk1 = sidx * 2 + 1;
#pragma unroll
      for (int h = 0; h < 8; ++h) {
        const int qo = h * 16;
        float d0 = 0.f, d1 = 0.f;
#pragma unroll
        for (int j = 0; j < 16; ++j) {
          float qv = (float)qkvb[r1 * QS + qo + j];
          d0 += qv * (float)qkvb[rk0 * QS + 128 + qo + j];
          d1 += qv * (float)qkvb[rk1 * QS + 128 + qo + j];
        }
        d0 *= 0.25f; d1 *= 0.25f;                         // 1/sqrt(DH)
        float mx = fmaxf(d0, d1);
        float e0 = __expf(d0 - mx), e1 = __expf(d1 - mx);
        float inv = 1.0f / (e0 + e1);
        e0 *= inv; e1 *= inv;
#pragma unroll
        for (int j = 0; j < 16; ++j) {
          float o = e0 * (float)qkvb[rk0 * QS + 256 + qo + j] +
                    e1 * (float)qkvb[rk1 * QS + 256 + qo + j];
          obuf[r1 * OS + qo + j] = (__bf16)o;
        }
      }
    }
    __syncthreads();

    // out-proj + residual: tok += o @ out_w^T + b
    for (int rt = 0; rt < 2; ++rt) {
      v16bf a4[4];
#pragma unroll
      for (int k4 = 0; k4 < 4; ++k4)
        a4[k4] = loadA_bf16(obuf + rt * 16 * OS, OS, k4 * 32, lane);
      for (int nt = 0; nt < 8; ++nt) {
        v8f acc = {};
#pragma unroll
        for (int k4 = 0; k4 < 4; ++k4)
          acc = wmma_bf16(a4[k4], loadB(oW, D, nt * 16, k4 * 32, lane), acc);
        const int n = nt * 16 + nl;
        const float bias = outB[li * D + n];
#pragma unroll
        for (int r = 0; r < 8; ++r)
          tokf[(rt * 16 + rb + r) * SF + n] += acc[r] + bias;
      }
    }
    __syncthreads();
    ln_row(tokf + lane * SF, tokb + lane * TBS, ln1w + li * D, ln1b + li * D);
    __syncthreads();

    // FFN up: h = gelu(tok @ ff1_w^T + b) -> hbuf[32][512] (bf16)
    for (int rt = 0; rt < 2; ++rt) {
      v16bf a4[4];
#pragma unroll
      for (int k4 = 0; k4 < 4; ++k4)
        a4[k4] = loadA_bf16(tokb + rt * 16 * TBS, TBS, k4 * 32, lane);
      for (int nt = 0; nt < 32; ++nt) {
        v8f acc = {};
#pragma unroll
        for (int k4 = 0; k4 < 4; ++k4)
          acc = wmma_bf16(a4[k4], loadB(f1W, D, nt * 16, k4 * 32, lane), acc);
        const int n = nt * 16 + nl;
        const float bias = ff1B[li * DFF + n];
#pragma unroll
        for (int r = 0; r < 8; ++r)
          hbuf[(rt * 16 + rb + r) * HS + n] = (__bf16)gelu_exact(acc[r] + bias);
      }
    }
    __syncthreads();

    // FFN down + residual: tok += h @ ff2_w^T + b   (K = 512)
    // k-outer / n-inner: each A fragment loaded once, reused for all 8 N tiles.
    for (int rt = 0; rt < 2; ++rt) {
      const __bf16* at = hbuf + rt * 16 * HS;
      v8f zero = {};
      v8f acc[8];
#pragma unroll
      for (int nt = 0; nt < 8; ++nt) acc[nt] = zero;
      for (int kb = 0; kb < DFF; kb += 32) {
        v16bf a = loadA_bf16(at, HS, kb, lane);
#pragma unroll
        for (int nt = 0; nt < 8; ++nt)
          acc[nt] = wmma_bf16(a, loadB(f2W, DFF, nt * 16, kb, lane), acc[nt]);
      }
#pragma unroll
      for (int nt = 0; nt < 8; ++nt) {
        const int n = nt * 16 + nl;
        const float bias = ff2B[li * D + n];
#pragma unroll
        for (int r = 0; r < 8; ++r)
          tokf[(rt * 16 + rb + r) * SF + n] += acc[nt][r] + bias;
      }
    }
    __syncthreads();
    ln_row(tokf + lane * SF, tokb + lane * TBS, ln2w + li * D, ln2b + li * D);
    __syncthreads();
  }

  // ---- final norm, mean-pool over T, cls_ln, gelu, classifier ----
  ln_row(tokf + lane * SF, tokb + lane * TBS, lnfw, lnfb);
  __syncthreads();
  if (lane < SPW) {
    const int s = lane;
    float* row = pooled + s * D;
    for (int d = 0; d < D; ++d)
      row[d] = 0.5f * (tokf[(2 * s) * SF + d] + tokf[(2 * s + 1) * SF + d]);
    float mean = 0.f;
    for (int d = 0; d < D; ++d) mean += row[d];
    mean *= (1.0f / D);
    float var = 0.f;
    for (int d = 0; d < D; ++d) { float t = row[d] - mean; var += t * t; }
    const float inv = rsqrtf(var * (1.0f / D) + 1e-5f);
    float o[NC];
#pragma unroll
    for (int c = 0; c < NC; ++c) o[c] = clsB[c];
    for (int d = 0; d < D; ++d) {
      float pv = (row[d] - mean) * inv * clnw[d] + clnb[d];
      pv = gelu_exact(pv);
#pragma unroll
      for (int c = 0; c < NC; ++c) o[c] += pv * clsW[c * D + d];
    }
#pragma unroll
    for (int c = 0; c < NC; ++c) out[(size_t)(s0 + s) * NC + c] = o[c];
  }
}

extern "C" void kernel_launch(void* const* d_in, const int* in_sizes, int n_in,
                              void* d_out, int out_size, void* d_ws, size_t ws_size,
                              hipStream_t stream) {
  (void)in_sizes; (void)n_in; (void)out_size; (void)ws_size;
  const float* x    = (const float*)d_in[0];
  const float* tpW  = (const float*)d_in[1];
  const float* tpB  = (const float*)d_in[2];
  const float* pos  = (const float*)d_in[3];
  const float* qkvW = (const float*)d_in[4];
  const float* qkvB = (const float*)d_in[5];
  const float* outW = (const float*)d_in[6];
  const float* outB = (const float*)d_in[7];
  const float* ln1w = (const float*)d_in[8];
  const float* ln1b = (const float*)d_in[9];
  const float* ff1W = (const float*)d_in[10];
  const float* ff1B = (const float*)d_in[11];
  const float* ff2W = (const float*)d_in[12];
  const float* ff2B = (const float*)d_in[13];
  const float* ln2w = (const float*)d_in[14];
  const float* ln2b = (const float*)d_in[15];
  const float* lnfw = (const float*)d_in[16];
  const float* lnfb = (const float*)d_in[17];
  const float* clnw = (const float*)d_in[18];
  const float* clnb = (const float*)d_in[19];
  const float* clsW = (const float*)d_in[20];
  const float* clsB = (const float*)d_in[21];
  float* out = (float*)d_out;

  __bf16* ws = (__bf16*)d_ws;
  auto cvt = [&](const float* s, __bf16* d, int n) {
    cvt_f32_bf16<<<(n + 255) / 256, 256, 0, stream>>>(s, d, n);
  };
  cvt(tpW,  ws + OFF_TP,  256 * 256);
  cvt(qkvW, ws + OFF_QKV, L * 3 * D * D);
  cvt(outW, ws + OFF_OUT, L * D * D);
  cvt(ff1W, ws + OFF_FF1, L * DFF * D);
  cvt(ff2W, ws + OFF_FF2, L * D * DFF);

  const int blocks = B_TOT / (WPB * SPW);               // 2048
  const size_t shmem = (size_t)WPB * WAVE_LDS;          // 118784 B
  vit_fused_kernel<<<blocks, WPB * 32, shmem, stream>>>(
      x, ws + OFF_TP, ws + OFF_QKV, ws + OFF_OUT, ws + OFF_FF1, ws + OFF_FF2,
      tpB, pos, qkvB, outB, ln1w, ln1b, ff1B, ff2B, ln2w, ln2b,
      lnfw, lnfb, clnw, clnb, clsW, clsB, out);
}
